// UpdateEdge_14190571946520
// MI455X (gfx1250) — matrix operand order repaired and checked by reading, not compiled, not run on hardware
//
#include <hip/hip_runtime.h>

typedef __attribute__((ext_vector_type(16))) __bf16 v16bf;
typedef __attribute__((ext_vector_type(8)))  float  v8f;

#define E_TOT  250000
#define E_ACT  200000
#define D_IR   240
#define LATD   128
#define OHD    128

// ---------------- arena float offsets (16-edge tile, 13184 floats = 52.7KB) ---
#define OF_IN0   0      // bf16 16x352  (2816 f)
#define OF_IP1   2816   // bf16 16x160  (1280 f)
#define OF_IM1   4096   // bf16 16x160  (1280 f)
#define OF_R2C4  5376   // bf16 16x64   (512 f)
#define OF_R2C0  5888   // bf16 16x64   (512 f)
#define OF_LAT   6400   // f32  16x128  (2048 f)  [live to end]
#define OF_LATB  8448   // bf16 16x128  (1024 f)  [live thru stage 5]
#define OF_G     9472   // f32  16x224  (3584 f)
#define OF_OM0   0      // f32  16x160
#define OF_OP1   2560   // f32  16x48
#define OF_ON1   3328   // f32  16x48
#define OF_OP2   4096   // f32  16x16
#define OF_ON2   4352   // f32  16x16
#define OF_Z0B   4608   // bf16 16x64   (512 f)
#define OF_Z1B   5120   // bf16 3x16x32 (768 f)
#define OF_Z2B   9472   // bf16 5x16x32 (1280 f)  [over dead gbuf]
#define OF_GATE  10752  // f32  16x48   (768 f)
#define OF_SCAL  0      // f32  16x64
#define OF_W     1024   // f32  16x112
#define OF_Z1P   2816   // f32  3x16x32 (1536 f)
#define OF_Z2P   4352   // f32  5x16x16 (1280 f)
#define OF_OHB   9472   // bf16 16x128  (1024 f)  [over dead z2b]
#define OF_TP    10752  // f32  16x112  (1792 f)  [over dead gate]
#define OF_CATA  1024   // bf16 16x192  (1536 f)  [over dead wbuf]
#define OF_HA    2560   // bf16 16x128  (1024 f)
#define OF_HB    3584   // bf16 16x128  (1024 f)
#define OF_CATB  10752  // bf16 16x256  (2048 f)  [over dead tp]
#define OF_IDX   13056

__device__ __forceinline__ v8f zero8() {
  v8f z;
#pragma unroll
  for (int i = 0; i < 8; ++i) z[i] = 0.f;
  return z;
}

// A fragment (16x32 bf16) from bf16 LDS rows; ISA 7.12.2 16-bit A layout.
__device__ __forceinline__ v16bf load_a_bf(const __bf16* base, int stride, int kt, int lane) {
  const __bf16* p = base + (lane & 15) * stride + kt * 32 + ((lane & 16) ? 8 : 0);
  v16bf a;
#pragma unroll
  for (int i = 0; i < 8; ++i) a[i] = p[i];
#pragma unroll
  for (int i = 0; i < 8; ++i) a[8 + i] = p[16 + i];
  return a;
}
__device__ __forceinline__ v16bf load_b(const __bf16* w, int KT, int nt, int kt, int lane) {
  return *(const v16bf*)(w + ((size_t)(nt * KT + kt) * 32 + lane) * 16);
}
__device__ __forceinline__ v8f wmma_bf16(v16bf a, v16bf b, v8f c) {
  return __builtin_amdgcn_wmma_f32_16x16x32_bf16(false, a, false, b, (short)0, c, false, false);
}
__device__ __forceinline__ v8f mm_bf(const __bf16* a, int astr, const void* wv, int KT, int nt, int lane) {
  const __bf16* w = (const __bf16*)wv;
  v8f acc = zero8();
  for (int kt = 0; kt < KT; ++kt)
    acc = wmma_bf16(load_a_bf(a, astr, kt, lane), load_b(w, KT, nt, kt, lane), acc);
  return acc;
}
// C/D layout: col n = lane&15, rows M = r + 8*(lane>=16)
__device__ __forceinline__ void store_c(float* base, int stride, int nt, int lane, v8f c) {
  int n = nt * 16 + (lane & 15);
  int mo = (lane & 16) ? 8 : 0;
#pragma unroll
  for (int r = 0; r < 8; ++r) base[(mo + r) * stride + n] = c[r];
}
__device__ __forceinline__ void store_c_bf(__bf16* base, int stride, int nt, int lane, v8f c) {
  int n = nt * 16 + (lane & 15);
  int mo = (lane & 16) ? 8 : 0;
#pragma unroll
  for (int r = 0; r < 8; ++r) base[(mo + r) * stride + n] = (__bf16)c[r];
}
__device__ __forceinline__ float sigmoidf(float x) { return 1.f / (1.f + __expf(-x)); }
__device__ __forceinline__ float siluf(float x) { return x * sigmoidf(x); }

struct KParams {
  const float *lat, *nf, *ef, *cut, *oh, *wD1, *wD2;
  const float *Gb, *P0b, *Eb, *lng, *lnb, *A1b, *A2b, *A3b, *B1b, *B2b, *B3b;
  const int *eidx, *act;
  float *out_edge, *out_lat;
  const void *W0, *W1r, *W1i, *W2r, *W2i, *Gw, *P0, *P1, *P2, *Ew;
  const void *A1, *A2, *A3, *B1, *B2, *B3, *O0, *O1, *O2;
};

// Pack f32 weight into bf16 WMMA B-fragment layout. tr=1: W stored [N][K] (x@W.T)
__global__ void pack_b_kernel(const float* __restrict__ W, void* dstv, int Kr, int Nr, int KT, int tr) {
  __bf16* dst = (__bf16*)dstv;
  int lane = threadIdx.x;
  int kt = blockIdx.x % KT;
  int nt = blockIdx.x / KT;
  int n = nt * 16 + (lane & 15);
  int kb = kt * 32 + ((lane & 16) ? 8 : 0);
  v16bf f;
#pragma unroll
  for (int h = 0; h < 16; ++h) {
    int k = kb + (h < 8 ? h : h + 8);
    float v = 0.f;
    if (k < Kr && n < Nr) v = tr ? W[(size_t)n * Kr + k] : W[(size_t)k * Nr + n];
    f[h] = (__bf16)v;
  }
  *(v16bf*)(dst + ((size_t)blockIdx.x * 32 + lane) * 16) = f;
}

__global__ void copy_latents_kernel(const float* __restrict__ src, float* __restrict__ dst, int n) {
  int i = blockIdx.x * 256 + threadIdx.x;
  if (i < n) dst[i] = src[i];
}

__global__ __launch_bounds__(128) void edge_update_kernel(KParams p) {
  __shared__ float sm[13184];
  const int tid = threadIdx.x;
  const int lane = tid & 31;
  const int wave = tid >> 5;
  const int tile = blockIdx.x;

  int* s_ae = (int*)&sm[OF_IDX];
  int* s_ec = (int*)&sm[OF_IDX + 16];
  int* s_en = (int*)&sm[OF_IDX + 32];
  float* s_cut = &sm[OF_IDX + 48];
  float* s_mu = &sm[OF_IDX + 64];
  float* s_rs = &sm[OF_IDX + 80];

  // prefetch the streamed per-edge rows for this tile (global_prefetch_b8)
  __builtin_prefetch(p.ef + (size_t)(tile * 16) * D_IR, 0, 0);
  __builtin_prefetch(p.oh + (size_t)(tile * 16) * OHD, 0, 0);
  __builtin_prefetch(p.wD1 + (size_t)(tile * 16) * 9, 0, 0);
  __builtin_prefetch(p.wD2 + (size_t)(tile * 16) * 25, 0, 0);

  if (tid < 16) {
    int row = tile * 16 + tid;
    int ae = p.act[row];
    s_ae[tid] = ae;
    s_ec[tid] = p.eidx[ae];
    s_en[tid] = p.eidx[E_TOT + ae];
    s_cut[tid] = p.cut[ae];
  }
  for (int i = tid; i < 6400; i += 128) sm[i] = 0.f;   // zero bf16 regions incl. K pads
  __syncthreads();

  __bf16* in0b = (__bf16*)&sm[OF_IN0];
  __bf16* ip1b = (__bf16*)&sm[OF_IP1];
  __bf16* im1b = (__bf16*)&sm[OF_IM1];
  __bf16* r2c4b = (__bf16*)&sm[OF_R2C4];
  __bf16* r2c0b = (__bf16*)&sm[OF_R2C0];
  float* latb = &sm[OF_LAT];
  __bf16* latbf = (__bf16*)&sm[OF_LATB];

  // ---- Stage 1: gather + forward Wigner rotation (fused) ----
  for (int i = tid; i < 16 * 192; i += 128) {       // x0 = [nc0|ef0|nn0]
    int e = i / 192, c = i % 192;
    float v;
    if (c < 64)       v = p.nf[(size_t)s_ec[e] * D_IR + c];
    else if (c < 128) v = p.ef[(size_t)(tile * 16 + e) * D_IR + (c - 64)];
    else              v = p.nf[(size_t)s_en[e] * D_IR + (c - 128)];
    in0b[e * 352 + c] = (__bf16)v;
  }
  for (int i = tid; i < 16 * 96; i += 128) {        // l=1: r1 = D1 @ x1
    int e = i / 96, u = i % 96;
    const float* src;
    if (u < 32)      src = p.nf + (size_t)s_ec[e] * D_IR + 64 + 3 * u;
    else if (u < 64) src = p.ef + (size_t)(tile * 16 + e) * D_IR + 64 + 3 * (u - 32);
    else             src = p.nf + (size_t)s_en[e] * D_IR + 64 + 3 * (u - 64);
    float a0 = src[0], a1 = src[1], a2 = src[2];
    const float* D = p.wD1 + (size_t)(tile * 16 + e) * 9;
    im1b[e * 160 + u]        = (__bf16)(D[0] * a0 + D[1] * a1 + D[2] * a2); // r1[:,:,0]
    in0b[e * 352 + 192 + u]  = (__bf16)(D[3] * a0 + D[4] * a1 + D[5] * a2); // r1[:,:,1]
    ip1b[e * 160 + u]        = (__bf16)(D[6] * a0 + D[7] * a1 + D[8] * a2); // r1[:,:,2]
  }
  for (int i = tid; i < 16 * 48; i += 128) {        // l=2: r2 = D2 @ x2
    int e = i / 48, u = i % 48;
    const float* src;
    if (u < 16)      src = p.nf + (size_t)s_ec[e] * D_IR + 160 + 5 * u;
    else if (u < 32) src = p.ef + (size_t)(tile * 16 + e) * D_IR + 160 + 5 * (u - 16);
    else             src = p.nf + (size_t)s_en[e] * D_IR + 160 + 5 * (u - 32);
    float x[5];
#pragma unroll
    for (int j = 0; j < 5; ++j) x[j] = src[j];
    const float* D = p.wD2 + (size_t)(tile * 16 + e) * 25;
    float r[5];
#pragma unroll
    for (int k2 = 0; k2 < 5; ++k2) {
      float s = 0.f;
#pragma unroll
      for (int j = 0; j < 5; ++j) s += D[5 * k2 + j] * x[j];
      r[k2] = s;
    }
    r2c0b[e * 64 + u]        = (__bf16)r[0];
    im1b[e * 160 + 96 + u]   = (__bf16)r[1];
    in0b[e * 352 + 288 + u]  = (__bf16)r[2];
    ip1b[e * 160 + 96 + u]   = (__bf16)r[3];
    r2c4b[e * 64 + u]        = (__bf16)r[4];
  }
  for (int i = tid; i < 16 * LATD; i += 128) {
    int e = i >> 7, c = i & 127;
    float v = p.lat[(size_t)s_ae[e] * LATD + c];
    latb[i] = v;
    latbf[i] = (__bf16)v;
  }
  __syncthreads();

  // ---- Stage 2: g = lat @ Gw + Gb ----
  float* gbuf = &sm[OF_G];
  for (int nt = wave; nt < 14; nt += 4) {
    v8f c = mm_bf(latbf, LATD, p.Gw, 4, nt, lane);
    float b = p.Gb[nt * 16 + (lane & 15)];
#pragma unroll
    for (int r = 0; r < 8; ++r) c[r] += b;
    store_c(gbuf, 224, nt, lane, c);
  }
  __syncthreads();

  // ---- Stage 3: W0 / W1 / W2 matmuls, gated ----
  v8f cres[3]; int ntsv[3]; int nres = 0;
  for (int nt = wave; nt < 10; nt += 4) {
    v8f c = mm_bf(in0b, 352, p.W0, 11, nt, lane);
    int n = nt * 16 + (lane & 15);
    int mo = (lane & 16) ? 8 : 0;
#pragma unroll
    for (int r = 0; r < 8; ++r) c[r] *= gbuf[(mo + r) * 224 + n];
    cres[nres] = c; ntsv[nres] = nt; ++nres;
  }
  v8f vop1 = zero8(), von1 = zero8(), vop2 = zero8(), von2 = zero8();
  if (wave < 3) {
    int nt = wave;
    v8f a1 = mm_bf(ip1b, 160, p.W1r, 5, nt, lane);
    v8f a2 = mm_bf(im1b, 160, p.W1i, 5, nt, lane);
    v8f a3 = mm_bf(ip1b, 160, p.W1i, 5, nt, lane);
    v8f a4 = mm_bf(im1b, 160, p.W1r, 5, nt, lane);
    int n = nt * 16 + (lane & 15);
    int mo = (lane & 16) ? 8 : 0;
#pragma unroll
    for (int r = 0; r < 8; ++r) {
      float gg = gbuf[(mo + r) * 224 + 160 + n];
      vop1[r] = (a1[r] - a2[r]) * gg;
      von1[r] = (a3[r] + a4[r]) * gg;
    }
  } else {
    v8f a1 = mm_bf(r2c4b, 64, p.W2r, 2, 0, lane);
    v8f a2 = mm_bf(r2c0b, 64, p.W2i, 2, 0, lane);
    v8f a3 = mm_bf(r2c4b, 64, p.W2i, 2, 0, lane);
    v8f a4 = mm_bf(r2c0b, 64, p.W2r, 2, 0, lane);
    int n = lane & 15;
    int mo = (lane & 16) ? 8 : 0;
#pragma unroll
    for (int r = 0; r < 8; ++r) {
      float gg = gbuf[(mo + r) * 224 + 208 + n];
      vop2[r] = (a1[r] - a2[r]) * gg;
      von2[r] = (a3[r] + a4[r]) * gg;
    }
  }
  __syncthreads();
  float* om0 = &sm[OF_OM0];
  float* op1 = &sm[OF_OP1];
  float* on1 = &sm[OF_ON1];
  float* op2 = &sm[OF_OP2];
  float* on2 = &sm[OF_ON2];
  for (int k = 0; k < nres; ++k) store_c(om0, 160, ntsv[k], lane, cres[k]);
  if (wave < 3) { store_c(op1, 48, wave, lane, vop1); store_c(on1, 48, wave, lane, von1); }
  else          { store_c(op2, 16, 0, lane, vop2);    store_c(on2, 16, 0, lane, von2); }
  __syncthreads();

  // ---- Stage 4: gates/silu + inverse Wigner -> z buffers (bf16 A-operands) ----
  __bf16* z0b = (__bf16*)&sm[OF_Z0B];
  __bf16* z1b = (__bf16*)&sm[OF_Z1B];
  __bf16* z2b = (__bf16*)&sm[OF_Z2B];
  float* gate = &sm[OF_GATE];
  for (int i = tid; i < 5 * 16 * 16; i += 128) {    // zero z2 K-pad (u=16..31)
    int m = i >> 8, r = i & 255, e = r >> 4, u = r & 15;
    z2b[(m * 16 + e) * 32 + 16 + u] = (__bf16)0.f;
  }
  for (int i = tid; i < 16 * 64; i += 128) {
    int e = i >> 6, c = i & 63;
    z0b[i] = (__bf16)siluf(om0[e * 160 + c]);
  }
  for (int i = tid; i < 16 * 48; i += 128) {
    int e = i / 48, u = i % 48;
    gate[i] = sigmoidf(om0[e * 160 + 64 + u]);
  }
  __syncthreads();
  for (int i = tid; i < 16 * 32; i += 128) {        // l=1 inverse rotation
    int e = i >> 5, u = i & 31;
    float y0 = on1[e * 48 + u];
    float y1 = om0[e * 160 + 112 + u];
    float y2 = op1[e * 48 + u];
    const float* D = p.wD1 + (size_t)(tile * 16 + e) * 9;
    float gg = gate[e * 48 + u];
    z1b[(0 * 16 + e) * 32 + u] = (__bf16)((D[0] * y0 + D[3] * y1 + D[6] * y2) * gg);
    z1b[(1 * 16 + e) * 32 + u] = (__bf16)((D[1] * y0 + D[4] * y1 + D[7] * y2) * gg);
    z1b[(2 * 16 + e) * 32 + u] = (__bf16)((D[2] * y0 + D[5] * y1 + D[8] * y2) * gg);
  }
  for (int i = tid; i < 16 * 16; i += 128) {        // l=2 inverse rotation
    int e = i >> 4, u = i & 15;
    float yj[5];
    yj[0] = on2[e * 16 + u];
    yj[1] = on1[e * 48 + 32 + u];
    yj[2] = om0[e * 160 + 144 + u];
    yj[3] = op1[e * 48 + 32 + u];
    yj[4] = op2[e * 16 + u];
    const float* D = p.wD2 + (size_t)(tile * 16 + e) * 25;
    float gg = gate[e * 48 + 32 + u];
#pragma unroll
    for (int m = 0; m < 5; ++m) {
      float s = 0.f;
#pragma unroll
      for (int j = 0; j < 5; ++j) s += D[5 * j + m] * yj[j];
      z2b[(m * 16 + e) * 32 + u] = (__bf16)(s * gg);
    }
  }
  __syncthreads();

  // ---- Stage 5: P projections + Ew ----
  v8f wres[2]; int wnts[2]; int nw = 0;
  for (int nt = wave; nt < 7; nt += 4) {
    v8f c = mm_bf(latbf, LATD, p.Ew, 4, nt, lane);
    float b = p.Eb[nt * 16 + (lane & 15)];
#pragma unroll
    for (int r = 0; r < 8; ++r) c[r] += b;
    wres[nw] = c; wnts[nw] = nt; ++nw;
  }
  v8f pres[5]; int np = 0;
  if (wave == 3) {
    for (int nt = 0; nt < 4; ++nt) {
      v8f c = mm_bf(z0b, 64, p.P0, 2, nt, lane);
      float b = p.P0b[nt * 16 + (lane & 15)];
#pragma unroll
      for (int r = 0; r < 8; ++r) c[r] = c[r] * 0.125f + b;   // /sqrt(64)
      pres[np++] = c;
    }
  } else if (wave == 0) {
    for (int m = 0; m < 5; ++m) {
      v8f c = mm_bf(z2b + m * 16 * 32, 32, p.P2, 1, 0, lane);
#pragma unroll
      for (int r = 0; r < 8; ++r) c[r] *= 0.25f;              // /sqrt(16)
      pres[np++] = c;
    }
  } else {
    const float invs = 0.17677669529663687f;                  // 1/sqrt(32)
    int t0 = (wave == 1) ? 0 : 3;
    for (int t = t0; t < t0 + 3; ++t) {
      v8f c = mm_bf(z1b + (t >> 1) * 16 * 32, 32, p.P1, 1, t & 1, lane);
#pragma unroll
      for (int r = 0; r < 8; ++r) c[r] *= invs;
      pres[np++] = c;
    }
  }
  __syncthreads();
  float* scal = &sm[OF_SCAL];
  float* wbuf = &sm[OF_W];
  float* z1p = &sm[OF_Z1P];
  float* z2p = &sm[OF_Z2P];
  for (int k = 0; k < nw; ++k) store_c(wbuf, 112, wnts[k], lane, wres[k]);
  if (wave == 3)      { for (int nt = 0; nt < 4; ++nt) store_c(scal, 64, nt, lane, pres[nt]); }
  else if (wave == 0) { for (int m = 0; m < 5; ++m)    store_c(z2p + m * 16 * 16, 16, 0, lane, pres[m]); }
  else {
    int t0 = (wave == 1) ? 0 : 3;
    for (int k = 0; k < 3; ++k) { int t = t0 + k; store_c(z1p + (t >> 1) * 16 * 32, 32, t & 1, lane, pres[k]); }
  }
  __bf16* ohbb = (__bf16*)&sm[OF_OHB];               // overwrites consumed z2b
  for (int i = tid; i < 16 * OHD; i += 128) {
    int e = i >> 7, c = i & 127;
    ohbb[i] = (__bf16)p.oh[(size_t)(tile * 16 + e) * OHD + c];
  }
  __syncthreads();

  // ---- Stage 6: O-projections (one_hot @ O?.T) ----
  float* tp = &sm[OF_TP];
  for (int j = wave; j < 7; j += 4) {
    const void* om; int ntm;
    if (j < 4)      { om = p.O0; ntm = j; }
    else if (j < 6) { om = p.O1; ntm = j - 4; }
    else            { om = p.O2; ntm = 0; }
    v8f c = mm_bf(ohbb, OHD, om, 4, ntm, lane);
    store_c(tp, 112, j, lane, c);
  }
  __syncthreads();

  // ---- Stage 7: edge_out assembly + store ----
  const float c_old = 0.8944271909999159f;   // rsqrt(1.25)
  const float c_new = 0.4472135954999579f;   // 0.5 * c_old
  const float s_oh = 0.08838834764831845f;   // 1/sqrt(128)
  for (int i = tid; i < 16 * D_IR; i += 128) {
    int e = i / 240, c = i % 240;
    float efv = p.ef[(size_t)(tile * 16 + e) * D_IR + c];
    float nev, tw;
    if (c < 64) { nev = scal[e * 64 + c] * wbuf[e * 112 + c]; tw = tp[e * 112 + c]; }
    else if (c < 160) {
      int q = c - 64, uu = q / 3, m = q % 3;
      nev = z1p[(m * 16 + e) * 32 + uu] * wbuf[e * 112 + 64 + uu];
      tw = tp[e * 112 + 64 + uu];
    } else {
      int q = c - 160, uu = q / 5, m = q % 5;
      nev = z2p[(m * 16 + e) * 16 + uu] * wbuf[e * 112 + 96 + uu];
      tw = tp[e * 112 + 96 + uu];
    }
    float eo = c_old * efv + c_new * nev;
    p.out_edge[(size_t)(tile * 16 + e) * D_IR + c] = eo * (1.f + tw * s_oh);
  }
  __syncthreads();

  // ---- Stage 8: latent path (LayerNorm + two 3-layer MLPs, bf16 intermediates) ----
  if (tid < 16) {
    float mu = 0.f;
    for (int c2 = 0; c2 < LATD; ++c2) mu += latb[tid * LATD + c2];
    mu *= (1.f / LATD);
    float var = 0.f;
    for (int c2 = 0; c2 < LATD; ++c2) { float d = latb[tid * LATD + c2] - mu; var += d * d; }
    var *= (1.f / LATD);
    s_mu[tid] = mu; s_rs[tid] = rsqrtf(var + 1e-5f);
  }
  __syncthreads();
  __bf16* catA = (__bf16*)&sm[OF_CATA];
  for (int i = tid; i < 16 * 192; i += 128) {
    int e = i / 192, c = i % 192;
    float v = (c < 128)
        ? (latb[e * LATD + c] - s_mu[e]) * s_rs[e] * p.lng[c] + p.lnb[c]
        : scal[e * 64 + (c - 128)];
    catA[i] = (__bf16)v;
  }
  __syncthreads();
  __bf16* hA = (__bf16*)&sm[OF_HA];
  __bf16* hB = (__bf16*)&sm[OF_HB];
  for (int nt = wave; nt < 8; nt += 4) {
    v8f c = mm_bf(catA, 192, p.A1, 6, nt, lane);
    float b = p.A1b[nt * 16 + (lane & 15)];
#pragma unroll
    for (int r = 0; r < 8; ++r) c[r] = siluf(c[r] + b);
    store_c_bf(hA, 128, nt, lane, c);
  }
  __syncthreads();
  for (int nt = wave; nt < 8; nt += 4) {
    v8f c = mm_bf(hA, 128, p.A2, 4, nt, lane);
    float b = p.A2b[nt * 16 + (lane & 15)];
#pragma unroll
    for (int r = 0; r < 8; ++r) c[r] = siluf(c[r] + b);
    store_c_bf(hB, 128, nt, lane, c);
  }
  __syncthreads();
  for (int nt = wave; nt < 8; nt += 4) {
    v8f c = mm_bf(hB, 128, p.A3, 4, nt, lane);
    float b = p.A3b[nt * 16 + (lane & 15)];
#pragma unroll
    for (int r = 0; r < 8; ++r) c[r] = c[r] + b;
    store_c_bf(hA, 128, nt, lane, c);
  }
  __syncthreads();
  __bf16* catB = (__bf16*)&sm[OF_CATB];
  for (int i = tid; i < 16 * 256; i += 128) {
    int e = i >> 8, c = i & 255;
    catB[i] = (c < 128) ? hA[e * 128 + c] : ohbb[e * 128 + (c - 128)];
  }
  __syncthreads();
  for (int nt = wave; nt < 8; nt += 4) {
    v8f c = mm_bf(catB, 256, p.B1, 8, nt, lane);
    float b = p.B1b[nt * 16 + (lane & 15)];
#pragma unroll
    for (int r = 0; r < 8; ++r) c[r] = siluf(c[r] + b);
    store_c_bf(hA, 128, nt, lane, c);
  }
  __syncthreads();
  for (int nt = wave; nt < 8; nt += 4) {
    v8f c = mm_bf(hA, 128, p.B2, 4, nt, lane);
    float b = p.B2b[nt * 16 + (lane & 15)];
#pragma unroll
    for (int r = 0; r < 8; ++r) c[r] = siluf(c[r] + b);
    store_c_bf(hB, 128, nt, lane, c);
  }
  __syncthreads();
  // B3: finish in registers, fuse cutoff/mix, store straight to global (coalesced rows)
  for (int nt = wave; nt < 8; nt += 4) {
    v8f c = mm_bf(hB, 128, p.B3, 4, nt, lane);
    int n = nt * 16 + (lane & 15);
    float b = p.B3b[n];
    int mo = (lane & 16) ? 8 : 0;
#pragma unroll
    for (int r = 0; r < 8; ++r) {
      int e = mo + r;
      float h = c[r] + b;
      float nl = s_cut[e] * h;
      p.out_lat[(size_t)s_ae[e] * LATD + n] = c_new * nl + c_old * latb[e * LATD + n];
    }
  }
}

extern "C" void kernel_launch(void* const* d_in, const int* in_sizes, int n_in,
                              void* d_out, int out_size, void* d_ws, size_t ws_size,
                              hipStream_t stream) {
  (void)in_sizes; (void)n_in; (void)out_size; (void)ws_size;
  struct PD { int idx, Kr, Nr, Kp, Np, tr; };
  static const PD pd[19] = {
    { 7, 336, 160, 352, 160, 1},  // W0
    { 8, 144,  48, 160,  48, 1},  // W1r
    { 9, 144,  48, 160,  48, 1},  // W1i
    {10,  48,  16,  64,  16, 1},  // W2r
    {11,  48,  16,  64,  16, 1},  // W2i
    {12, 128, 224, 128, 224, 0},  // Gw
    {14,  64,  64,  64,  64, 1},  // P0
    {16,  32,  32,  32,  32, 1},  // P1
    {17,  16,  16,  32,  16, 1},  // P2
    {18, 128, 112, 128, 112, 0},  // Ew
    {22, 192, 128, 192, 128, 0},  // A1
    {24, 128, 128, 128, 128, 0},  // A2
    {26, 128, 128, 128, 128, 0},  // A3
    {28, 256, 128, 256, 128, 0},  // B1
    {30, 128, 128, 128, 128, 0},  // B2
    {32, 128, 128, 128, 128, 0},  // B3
    {34, 128,  64, 128,  64, 1},  // O0 (stored [64][128])
    {35, 128,  32, 128,  32, 1},  // O1
    {36, 128,  16, 128,  16, 1},  // O2
  };
  const void* wp[19];
  size_t off = 0;  // in bf16 elements
  for (int i = 0; i < 19; ++i) {
    int KT = pd[i].Kp / 32, NT = pd[i].Np / 16;
    void* dst = (void*)((char*)d_ws + off * 2);
    wp[i] = dst;
    pack_b_kernel<<<KT * NT, 32, 0, stream>>>((const float*)d_in[pd[i].idx], dst,
                                              pd[i].Kr, pd[i].Nr, KT, pd[i].tr);
    off += (size_t)pd[i].Kp * pd[i].Np;
  }

  float* out_edge = (float*)d_out;
  float* out_lat = (float*)d_out + (size_t)E_ACT * D_IR;
  copy_latents_kernel<<<(E_TOT * LATD + 255) / 256, 256, 0, stream>>>(
      (const float*)d_in[0], out_lat, E_TOT * LATD);

  KParams P;
  P.lat = (const float*)d_in[0];  P.nf  = (const float*)d_in[1];
  P.ef  = (const float*)d_in[2];  P.cut = (const float*)d_in[3];
  P.oh  = (const float*)d_in[4];  P.wD1 = (const float*)d_in[5];
  P.wD2 = (const float*)d_in[6];
  P.Gb  = (const float*)d_in[13]; P.P0b = (const float*)d_in[15];
  P.Eb  = (const float*)d_in[19]; P.lng = (const float*)d_in[20];
  P.lnb = (const float*)d_in[21];
  P.A1b = (const float*)d_in[23]; P.A2b = (const float*)d_in[25];
  P.A3b = (const float*)d_in[27];
  P.B1b = (const float*)d_in[29]; P.B2b = (const float*)d_in[31];
  P.B3b = (const float*)d_in[33];
  P.eidx = (const int*)d_in[37];  P.act = (const int*)d_in[38];
  P.out_edge = out_edge; P.out_lat = out_lat;
  P.W0 = wp[0]; P.W1r = wp[1]; P.W1i = wp[2]; P.W2r = wp[3]; P.W2i = wp[4];
  P.Gw = wp[5]; P.P0 = wp[6]; P.P1 = wp[7]; P.P2 = wp[8]; P.Ew = wp[9];
  P.A1 = wp[10]; P.A2 = wp[11]; P.A3 = wp[12];
  P.B1 = wp[13]; P.B2 = wp[14]; P.B3 = wp[15];
  P.O0 = wp[16]; P.O1 = wp[17]; P.O2 = wp[18];

  edge_update_kernel<<<E_ACT / 16, 128, 0, stream>>>(P);
}
